// RNN_31662498906734
// MI455X (gfx1250) — compile-verified, matrix operand order checked by reading, not run on hardware
//
#include <hip/hip_runtime.h>

// Problem constants (match the reference)
#define LNUM 4
#define BDIM 32
#define TDIM 2048
#define DDIM 512
#define HDIM 512

typedef __attribute__((ext_vector_type(16))) __bf16        v16bf;
typedef __attribute__((ext_vector_type(8)))  float         v8f;
typedef __attribute__((ext_vector_type(4)))  unsigned int  u32x4;

// CDNA5 16-bit A/B operand layout (ISA 7.12.2): lane L holds row (L&15),
// K = {hi*8 .. hi*8+7} in VGPR0-3 and K = {hi*8+16 .. hi*8+23} in VGPR4-7,
// hi = L>>4.  Both chunks are K-contiguous -> two b128 loads per lane.
union BF16x16 {
  v16bf v;
  u32x4 q[2];
};

__device__ __forceinline__ float halfsum16(float v) {
  // sum across the 16 lanes of a half-wave (wave32)
  v += __shfl_xor(v, 1, 32);
  v += __shfl_xor(v, 2, 32);
  v += __shfl_xor(v, 4, 32);
  v += __shfl_xor(v, 8, 32);
  return v;
}

__device__ __forceinline__ v8f wmma_bf16(const BF16x16& a, const BF16x16& b, v8f c) {
  // D(16x16,f32) = A(16x32,bf16) * B(32x16,bf16) + C
  return __builtin_amdgcn_wmma_f32_16x16x32_bf16(false, a.v, false, b.v,
                                                 (short)0, c, false, false);
}

// Async global->LDS copy of one 16B chunk (CDNA5 GLOBAL_LOAD_ASYNC_TO_LDS,
// GV mode; tracked by ASYNCcnt).  ldsAddr = low 32 bits of the flat
// shared-aperture address (ISA 10.2).
__device__ __forceinline__ void async_ldst_b128(unsigned ldsAddr, const void* gaddr) {
  asm volatile("global_load_async_to_lds_b128 %0, %1, off"
               :: "v"(ldsAddr), "v"((unsigned long long)(size_t)gaddr)
               : "memory");
}
__device__ __forceinline__ void wait_asynccnt0() {
  asm volatile("s_wait_asynccnt 0" ::: "memory");
}

// ---------------------------------------------------------------------------
// fp32 -> bf16 weight conversion (weights then stay L2-resident)
// ---------------------------------------------------------------------------
__global__ void f32_to_bf16(const float* __restrict__ src,
                            __bf16* __restrict__ dst, int n) {
  int i = blockIdx.x * blockDim.x + threadIdx.x;
  if (i < n) dst[i] = (__bf16)src[i];
}

// ---------------------------------------------------------------------------
// Split path, phase 2a (fully parallel): ziLn[l,t] = LN(W_ih[l] @ x_t + b_ih)
// for every (l,t).  Removes the entire input projection + its LayerNorm from
// the serial recurrence.  One WG per (l,t): 512 threads = 16 waves, each
// wave owns col tiles {w, w+16} x 2 row tiles.
// ---------------------------------------------------------------------------
__global__ __launch_bounds__(512)
void zi_ln_precompute(const float* __restrict__ x,     // [B,T,D]
                      const __bf16* __restrict__ Wih,  // [L,H,D] bf16
                      const float* __restrict__ bih,   // [L,H]
                      const float* __restrict__ gih,   // [L,H]
                      const float* __restrict__ beih,  // [L,H]
                      __bf16* __restrict__ ziLn)       // [L,T,B,H] bf16
{
  __shared__ __bf16 sx[BDIM * DDIM];   // x_t as bf16, 32 KB
  __shared__ float  red[64];           // row sum / sumsq

  const int l    = blockIdx.x >> 11;   // T = 2048 = 2^11
  const int t    = blockIdx.x & (TDIM - 1);
  const int tid  = threadIdx.x;
  const int wave = tid >> 5;           // 0..15
  const int lane = tid & 31;
  const int hl   = lane & 15;
  const int hi   = lane >> 4;
  const int col0 = wave * 16 + hl;         // first column tile
  const int col1 = (wave + 16) * 16 + hl;  // second column tile

  if (tid < 64) red[tid] = 0.0f;
  for (int i = tid; i < BDIM * DDIM; i += 512) {
    const int b = i >> 9, d = i & 511;
    sx[i] = (__bf16)x[(size_t)b * TDIM * DDIM + (size_t)t * DDIM + d];
  }

  const float bias0 = bih[l * HDIM + col0], bias1 = bih[l * HDIM + col1];
  const __bf16* Brow0 = Wih + (size_t)(l * HDIM + col0) * DDIM + hi * 8;
  const __bf16* Brow1 = Wih + (size_t)(l * HDIM + col1) * DDIM + hi * 8;

  __syncthreads();

  v8f a00, a01, a10, a11;   // [row tile][col tile]
  #pragma unroll
  for (int r = 0; r < 8; ++r) { a00[r] = bias0; a01[r] = bias1; a10[r] = bias0; a11[r] = bias1; }

  #pragma unroll 4
  for (int kt = 0; kt < 16; ++kt) {
    const int k0 = kt * 32;
    BF16x16 B0, B1, Am0, Am1;
    B0.q[0] = *(const u32x4*)(Brow0 + k0);
    B0.q[1] = *(const u32x4*)(Brow0 + k0 + 16);
    B1.q[0] = *(const u32x4*)(Brow1 + k0);
    B1.q[1] = *(const u32x4*)(Brow1 + k0 + 16);
    const __bf16* xr0 = sx + (0 * 16 + hl) * 512 + hi * 8 + k0;
    const __bf16* xr1 = sx + (1 * 16 + hl) * 512 + hi * 8 + k0;
    Am0.q[0] = *(const u32x4*)(xr0); Am0.q[1] = *(const u32x4*)(xr0 + 16);
    Am1.q[0] = *(const u32x4*)(xr1); Am1.q[1] = *(const u32x4*)(xr1 + 16);
    a00 = wmma_bf16(Am0, B0, a00);
    a01 = wmma_bf16(Am0, B1, a01);
    a10 = wmma_bf16(Am1, B0, a10);
    a11 = wmma_bf16(Am1, B1, a11);
  }

  // per-row LN stats across all 512 columns (both col tiles contribute)
  #pragma unroll
  for (int r = 0; r < 8; ++r) {
    float v, s1, s2;
    v = a00[r]; s1 = halfsum16(v); s2 = halfsum16(v * v);
    if (hl == 0) { const int row = hi * 8 + r;      atomicAdd(&red[row], s1); atomicAdd(&red[32 + row], s2); }
    v = a01[r]; s1 = halfsum16(v); s2 = halfsum16(v * v);
    if (hl == 0) { const int row = hi * 8 + r;      atomicAdd(&red[row], s1); atomicAdd(&red[32 + row], s2); }
    v = a10[r]; s1 = halfsum16(v); s2 = halfsum16(v * v);
    if (hl == 0) { const int row = 16 + hi * 8 + r; atomicAdd(&red[row], s1); atomicAdd(&red[32 + row], s2); }
    v = a11[r]; s1 = halfsum16(v); s2 = halfsum16(v * v);
    if (hl == 0) { const int row = 16 + hi * 8 + r; atomicAdd(&red[row], s1); atomicAdd(&red[32 + row], s2); }
  }
  __syncthreads();

  if (tid < 32) {
    const float mu  = red[tid] * (1.0f / 512.0f);
    const float var = red[32 + tid] * (1.0f / 512.0f) - mu * mu;
    red[tid]      = mu;
    red[32 + tid] = rsqrtf(var + 1e-5f);
  }
  __syncthreads();

  const float g0 = gih[l * HDIM + col0],  g1 = gih[l * HDIM + col1];
  const float e0 = beih[l * HDIM + col0], e1 = beih[l * HDIM + col1];
  __bf16* dst = ziLn + (size_t)(l * TDIM + t) * BDIM * HDIM;
  #pragma unroll
  for (int r = 0; r < 8; ++r) {
    const int row0 = hi * 8 + r;
    const int row1 = 16 + hi * 8 + r;
    dst[(size_t)row0 * HDIM + col0] = (__bf16)((a00[r] - red[row0]) * red[32 + row0] * g0 + e0);
    dst[(size_t)row0 * HDIM + col1] = (__bf16)((a01[r] - red[row0]) * red[32 + row0] * g1 + e1);
    dst[(size_t)row1 * HDIM + col0] = (__bf16)((a10[r] - red[row1]) * red[32 + row1] * g0 + e0);
    dst[(size_t)row1 * HDIM + col1] = (__bf16)((a11[r] - red[row1]) * red[32 + row1] * g1 + e1);
  }
}

// ---------------------------------------------------------------------------
// Split path, phase 2b (serial): h = tanh(ziLn[l,t] + LN(W_hh[l] @ h + b_hh))
// One WG per independent layer chain; 1024 threads = 32 waves, wave w owns
// columns [16w,16w+16).  LN(zi) for step t+1 is pulled into the alternate
// LDS buffer with CDNA5 async global->LDS loads while step t computes, so
// the copy never sits on the serial critical path.
// ---------------------------------------------------------------------------
#define ZBUF (BDIM * HDIM)   // elements per z buffer

__global__ __launch_bounds__(1024)
void rnn_rec_pre(const __bf16* __restrict__ ziLn,   // [L,T,B,H] bf16
                 const __bf16* __restrict__ Whh,    // [L,H,H] bf16
                 const float* __restrict__ bhh,     // [L,H]
                 const float* __restrict__ ghh,     // [L,H]
                 const float* __restrict__ behh,    // [L,H]
                 __bf16* __restrict__ h3,           // [T,B,H] staging (layer 3)
                 float* __restrict__ out)           // d_out
{
  extern __shared__ char smem[];
  __bf16* sh_h = (__bf16*)smem;                          // 32x512 bf16 = 32 KB
  __bf16* sh_z = (__bf16*)(smem + BDIM * HDIM * 2);      // 2 x 32 KB (double buffer)
  float*  red  = (float*)(smem + 3 * BDIM * HDIM * 2);   // 64 f32

  const int l    = blockIdx.x;
  const int tid  = threadIdx.x;
  const int wave = tid >> 5;
  const int lane = tid & 31;
  const int hl   = lane & 15;
  const int hi   = lane >> 4;
  const int col  = wave * 16 + hl;

  for (int i = tid; i < BDIM * HDIM; i += 1024) sh_h[i] = (__bf16)0.0f;

  const float bias_h = bhh [l * HDIM + col];
  const float gh     = ghh [l * HDIM + col];
  const float beh    = behh[l * HDIM + col];
  const __bf16* WhhRow = Whh + (size_t)(l * HDIM + col) * HDIM + hi * 8;
  const __bf16* zBase  = ziLn + (size_t)l * TDIM * BDIM * HDIM;

  // preload z[0] into buffer 0 (plain copy; only once)
  {
    const u32x4* src = (const u32x4*)zBase;
    u32x4* dst = (u32x4*)sh_z;
    #pragma unroll
    for (int i = 0; i < 2; ++i) dst[tid + i * 1024] = src[tid + i * 1024];
  }
  __syncthreads();

  for (int t = 0; t < TDIM; ++t) {
    const __bf16* zcur = sh_z + (t & 1) * ZBUF;
    __bf16*       znxt = sh_z + ((t + 1) & 1) * ZBUF;

    if (tid < 64) red[tid] = 0.0f;

    // async-copy next step's LN(zi) into the alternate buffer (2 x b128/lane)
    if (t + 1 < TDIM) {
      const char* gsrc = (const char*)(zBase + (size_t)(t + 1) * BDIM * HDIM);
      const unsigned ldsBase = (unsigned)(size_t)(void*)znxt;
      #pragma unroll
      for (int i = 0; i < 2; ++i) {
        const int c = tid + i * 1024;           // 2048 chunks of 16 B = 32 KB
        async_ldst_b128(ldsBase + c * 16, gsrc + c * 16);
      }
    }
    // prefetch t+2 into L2 so next step's async loads hit cache
    if (t + 2 < TDIM && tid < 512) {
      __builtin_prefetch(zBase + (size_t)(t + 2) * BDIM * HDIM + tid * 32, 0, 1);
    }
    __syncthreads();   // red zeroed; h from previous step visible

    // zh = h * W_hh^T + b_hh
    v8f zh0, zh1;
    #pragma unroll
    for (int r = 0; r < 8; ++r) { zh0[r] = bias_h; zh1[r] = bias_h; }

    #pragma unroll 4
    for (int kt = 0; kt < 16; ++kt) {
      const int k0 = kt * 32;
      BF16x16 Bh, Ah0, Ah1;
      Bh.q[0] = *(const u32x4*)(WhhRow + k0);
      Bh.q[1] = *(const u32x4*)(WhhRow + k0 + 16);
      const __bf16* hr0 = sh_h + (0 * 16 + hl) * 512 + hi * 8 + k0;
      const __bf16* hr1 = sh_h + (1 * 16 + hl) * 512 + hi * 8 + k0;
      Ah0.q[0] = *(const u32x4*)(hr0); Ah0.q[1] = *(const u32x4*)(hr0 + 16);
      Ah1.q[0] = *(const u32x4*)(hr1); Ah1.q[1] = *(const u32x4*)(hr1 + 16);
      zh0 = wmma_bf16(Ah0, Bh, zh0);
      zh1 = wmma_bf16(Ah1, Bh, zh1);
    }

    // LN(zh) stats
    #pragma unroll
    for (int r = 0; r < 8; ++r) {
      float v, s1, s2;
      v = zh0[r]; s1 = halfsum16(v); s2 = halfsum16(v * v);
      if (hl == 0) { const int row = hi * 8 + r;      atomicAdd(&red[row], s1); atomicAdd(&red[32 + row], s2); }
      v = zh1[r]; s1 = halfsum16(v); s2 = halfsum16(v * v);
      if (hl == 0) { const int row = 16 + hi * 8 + r; atomicAdd(&red[row], s1); atomicAdd(&red[32 + row], s2); }
    }
    __syncthreads();

    if (tid < 32) {
      const float mu  = red[tid] * (1.0f / 512.0f);
      const float var = red[32 + tid] * (1.0f / 512.0f) - mu * mu;
      red[tid]      = mu;
      red[32 + tid] = rsqrtf(var + 1e-5f);
    }
    __syncthreads();

    #pragma unroll
    for (int r = 0; r < 8; ++r) {
      const int row0 = hi * 8 + r;
      const int row1 = 16 + hi * 8 + r;
      const float hn0 = tanhf((float)zcur[row0 * 512 + col] +
                              (zh0[r] - red[row0]) * red[32 + row0] * gh + beh);
      const float hn1 = tanhf((float)zcur[row1 * 512 + col] +
                              (zh1[r] - red[row1]) * red[32 + row1] * gh + beh);
      sh_h[row0 * 512 + col] = (__bf16)hn0;
      sh_h[row1 * 512 + col] = (__bf16)hn1;
      if (l == LNUM - 1) {
        h3[((size_t)t * BDIM + row0) * HDIM + col] = (__bf16)hn0;
        h3[((size_t)t * BDIM + row1) * HDIM + col] = (__bf16)hn1;
      }
      if (t == TDIM - 1) {
        out[(size_t)BDIM * TDIM * DDIM + ((size_t)l * BDIM + row0) * HDIM + col] = hn0;
        out[(size_t)BDIM * TDIM * DDIM + ((size_t)l * BDIM + row1) * HDIM + col] = hn1;
      }
    }
    if (t + 1 < TDIM) wait_asynccnt0();   // znxt fully landed in LDS
    __syncthreads();
  }
}

// ---------------------------------------------------------------------------
// Fallback (small workspace): fully fused recurrence (zi + zh per step).
// Partial unroll keeps weight operands out of long live ranges (no spills).
// ---------------------------------------------------------------------------
__global__ __launch_bounds__(1024)
void rnn_rec_fused(const float* __restrict__ x,
                   const __bf16* __restrict__ Wih,
                   const float* __restrict__ bih,
                   const float* __restrict__ gih,
                   const float* __restrict__ beih,
                   const __bf16* __restrict__ Whh,
                   const float* __restrict__ bhh,
                   const float* __restrict__ ghh,
                   const float* __restrict__ behh,
                   __bf16* __restrict__ h3,
                   float* __restrict__ out)
{
  extern __shared__ char smem[];
  __bf16* sh_h = (__bf16*)smem;
  __bf16* sh_x = (__bf16*)(smem + BDIM * HDIM * 2);
  float*  red  = (float*)(smem + 2 * BDIM * HDIM * 2);  // 128 f32

  const int l    = blockIdx.x;
  const int tid  = threadIdx.x;
  const int wave = tid >> 5;
  const int lane = tid & 31;
  const int hl   = lane & 15;
  const int hi   = lane >> 4;
  const int col  = wave * 16 + hl;

  for (int i = tid; i < BDIM * HDIM; i += 1024) sh_h[i] = (__bf16)0.0f;

  const float bias_i = bih [l * HDIM + col];
  const float bias_h = bhh [l * HDIM + col];
  const float gi  = gih [l * HDIM + col];
  const float bei = beih[l * HDIM + col];
  const float gh  = ghh [l * HDIM + col];
  const float beh = behh[l * HDIM + col];

  const __bf16* WihRow = Wih + (size_t)(l * HDIM + col) * DDIM + hi * 8;
  const __bf16* WhhRow = Whh + (size_t)(l * HDIM + col) * HDIM + hi * 8;

  __syncthreads();

  for (int t = 0; t < TDIM; ++t) {
    if (tid < 128) red[tid] = 0.0f;
    for (int i = tid; i < BDIM * DDIM; i += 1024) {
      const int b = i >> 9, d = i & 511;
      sh_x[i] = (__bf16)x[(size_t)b * TDIM * DDIM + (size_t)t * DDIM + d];
    }
    if (t + 1 < TDIM) {
      const int pb = tid >> 5, pl = tid & 31;
      __builtin_prefetch(x + (size_t)pb * TDIM * DDIM + (size_t)(t + 1) * DDIM + pl * 16, 0, 1);
    }
    __syncthreads();

    v8f zi0, zi1, zh0, zh1;
    #pragma unroll
    for (int r = 0; r < 8; ++r) { zi0[r] = bias_i; zi1[r] = bias_i; zh0[r] = bias_h; zh1[r] = bias_h; }

    #pragma unroll 2
    for (int kt = 0; kt < 16; ++kt) {
      const int k0 = kt * 32;
      BF16x16 Bi, Bh, Ax0, Ax1, Ah0, Ah1;
      Bi.q[0] = *(const u32x4*)(WihRow + k0);
      Bi.q[1] = *(const u32x4*)(WihRow + k0 + 16);
      Bh.q[0] = *(const u32x4*)(WhhRow + k0);
      Bh.q[1] = *(const u32x4*)(WhhRow + k0 + 16);
      const __bf16* xr0 = sh_x + (0 * 16 + hl) * 512 + hi * 8 + k0;
      const __bf16* xr1 = sh_x + (1 * 16 + hl) * 512 + hi * 8 + k0;
      const __bf16* hr0 = sh_h + (0 * 16 + hl) * 512 + hi * 8 + k0;
      const __bf16* hr1 = sh_h + (1 * 16 + hl) * 512 + hi * 8 + k0;
      Ax0.q[0] = *(const u32x4*)(xr0); Ax0.q[1] = *(const u32x4*)(xr0 + 16);
      Ax1.q[0] = *(const u32x4*)(xr1); Ax1.q[1] = *(const u32x4*)(xr1 + 16);
      Ah0.q[0] = *(const u32x4*)(hr0); Ah0.q[1] = *(const u32x4*)(hr0 + 16);
      Ah1.q[0] = *(const u32x4*)(hr1); Ah1.q[1] = *(const u32x4*)(hr1 + 16);
      zi0 = wmma_bf16(Ax0, Bi, zi0);
      zi1 = wmma_bf16(Ax1, Bi, zi1);
      zh0 = wmma_bf16(Ah0, Bh, zh0);
      zh1 = wmma_bf16(Ah1, Bh, zh1);
    }

    #pragma unroll
    for (int r = 0; r < 8; ++r) {
      float v, s1, s2;
      v = zi0[r]; s1 = halfsum16(v); s2 = halfsum16(v * v);
      if (hl == 0) { const int row = hi * 8 + r;      atomicAdd(&red[row],      s1); atomicAdd(&red[32 + row], s2); }
      v = zi1[r]; s1 = halfsum16(v); s2 = halfsum16(v * v);
      if (hl == 0) { const int row = 16 + hi * 8 + r; atomicAdd(&red[row],      s1); atomicAdd(&red[32 + row], s2); }
      v = zh0[r]; s1 = halfsum16(v); s2 = halfsum16(v * v);
      if (hl == 0) { const int row = hi * 8 + r;      atomicAdd(&red[64 + row], s1); atomicAdd(&red[96 + row], s2); }
      v = zh1[r]; s1 = halfsum16(v); s2 = halfsum16(v * v);
      if (hl == 0) { const int row = 16 + hi * 8 + r; atomicAdd(&red[64 + row], s1); atomicAdd(&red[96 + row], s2); }
    }
    __syncthreads();

    if (tid < 64) {
      const int typ = tid >> 5, row = tid & 31;
      const float mu  = red[typ * 64 + row] * (1.0f / 512.0f);
      const float var = red[typ * 64 + 32 + row] * (1.0f / 512.0f) - mu * mu;
      red[typ * 64 + row]      = mu;
      red[typ * 64 + 32 + row] = rsqrtf(var + 1e-5f);
    }
    __syncthreads();

    #pragma unroll
    for (int r = 0; r < 8; ++r) {
      const int row0 = hi * 8 + r;
      const int row1 = 16 + hi * 8 + r;
      const float hn0 = tanhf((zi0[r] - red[row0]) * red[32 + row0] * gi + bei +
                              (zh0[r] - red[64 + row0]) * red[96 + row0] * gh + beh);
      const float hn1 = tanhf((zi1[r] - red[row1]) * red[32 + row1] * gi + bei +
                              (zh1[r] - red[64 + row1]) * red[96 + row1] * gh + beh);
      sh_h[row0 * 512 + col] = (__bf16)hn0;
      sh_h[row1 * 512 + col] = (__bf16)hn1;
      if (l == LNUM - 1) {
        h3[((size_t)t * BDIM + row0) * HDIM + col] = (__bf16)hn0;
        h3[((size_t)t * BDIM + row1) * HDIM + col] = (__bf16)hn1;
      }
      if (t == TDIM - 1) {
        out[(size_t)BDIM * TDIM * DDIM + ((size_t)l * BDIM + row0) * HDIM + col] = hn0;
        out[(size_t)BDIM * TDIM * DDIM + ((size_t)l * BDIM + row1) * HDIM + col] = hn1;
      }
    }
    __syncthreads();
  }
}

// ---------------------------------------------------------------------------
// Phase 3: output head: out[b,t,:] = h3[t,b,:] @ W_ho^T + b_ho + x[b,t,:]
// ---------------------------------------------------------------------------
__global__ __launch_bounds__(256)
void out_gemm(const __bf16* __restrict__ h3,   // [T*B, H] bf16
              const __bf16* __restrict__ Who,  // [D, H]  bf16
              const float* __restrict__ bho,   // [D]
              const float* __restrict__ x,     // [B,T,D]
              float* __restrict__ out)         // [B,T,D]
{
  const int rt   = blockIdx.x;          // 16-row tile
  const int wave = threadIdx.x >> 5;    // 0..7
  const int lane = threadIdx.x & 31;
  const int hl   = lane & 15;
  const int hi   = lane >> 4;

  const __bf16* Arow = h3 + ((size_t)rt * 16 + hl) * HDIM + hi * 8;
  v8f acc[4] = {};

  #pragma unroll 4
  for (int kt = 0; kt < 16; ++kt) {
    const int k0 = kt * 32;
    BF16x16 A;
    A.q[0] = *(const u32x4*)(Arow + k0);
    A.q[1] = *(const u32x4*)(Arow + k0 + 16);
    #pragma unroll
    for (int j = 0; j < 4; ++j) {
      const int n = wave + 8 * j;
      const __bf16* Brow = Who + (size_t)(n * 16 + hl) * HDIM + hi * 8 + k0;
      BF16x16 Bv;
      Bv.q[0] = *(const u32x4*)(Brow);
      Bv.q[1] = *(const u32x4*)(Brow + 16);
      acc[j] = wmma_bf16(A, Bv, acc[j]);
    }
  }

  #pragma unroll
  for (int j = 0; j < 4; ++j) {
    const int col = (wave + 8 * j) * 16 + hl;
    const float bo = bho[col];
    #pragma unroll
    for (int r = 0; r < 8; ++r) {
      const int rg = rt * 16 + hi * 8 + r;      // global row = t*32 + b
      const int b = rg & 31, t = rg >> 5;
      const size_t idx = (size_t)b * TDIM * DDIM + (size_t)t * DDIM + col;
      out[idx] = acc[j][r] + bo + x[idx];
    }
  }
}

// ---------------------------------------------------------------------------
extern "C" void kernel_launch(void* const* d_in, const int* in_sizes, int n_in,
                              void* d_out, int out_size, void* d_ws, size_t ws_size,
                              hipStream_t stream) {
  (void)in_sizes; (void)n_in; (void)out_size;

  const float* x     = (const float*)d_in[0];
  const float* W_ih  = (const float*)d_in[1];
  const float* b_ih  = (const float*)d_in[2];
  const float* g_ih  = (const float*)d_in[3];
  const float* be_ih = (const float*)d_in[4];
  const float* W_hh  = (const float*)d_in[5];
  const float* b_hh  = (const float*)d_in[6];
  const float* g_hh  = (const float*)d_in[7];
  const float* be_hh = (const float*)d_in[8];
  const float* W_ho  = (const float*)d_in[9];
  const float* b_ho  = (const float*)d_in[10];
  float* out = (float*)d_out;

  // Workspace layout:
  //   [0, 2 MB)          W_ih bf16
  //   [2 MB, 4 MB)       W_hh bf16
  //   [4 MB, 4.5 MB)     W_ho bf16
  //   [4.5 MB, ~69 MB)   h3   bf16  [T,B,H]
  //   [~69 MB, ~325 MB)  ziLn bf16  [L,T,B,H]   (split path only)
  char* ws = (char*)d_ws;
  __bf16* wih  = (__bf16*)(ws);
  __bf16* whh  = (__bf16*)(ws + (size_t)2097152);
  __bf16* who  = (__bf16*)(ws + (size_t)4194304);
  __bf16* h3   = (__bf16*)(ws + (size_t)4718592);
  __bf16* ziLn = (__bf16*)(ws + (size_t)71827456);
  const size_t wsNeedSplit = (size_t)71827456 + (size_t)LNUM * TDIM * BDIM * HDIM * 2;

  const int nWih = LNUM * HDIM * DDIM;
  const int nWhh = LNUM * HDIM * HDIM;
  const int nWho = DDIM * HDIM;
  f32_to_bf16<<<(nWih + 255) / 256, 256, 0, stream>>>(W_ih, wih, nWih);
  f32_to_bf16<<<(nWhh + 255) / 256, 256, 0, stream>>>(W_hh, whh, nWhh);
  f32_to_bf16<<<(nWho + 255) / 256, 256, 0, stream>>>(W_ho, who, nWho);

  if (ws_size >= wsNeedSplit) {
    // Split path: hoist the h-independent half of every step out of the
    // recurrence into a fully parallel WMMA pass.
    zi_ln_precompute<<<LNUM * TDIM, 512, 0, stream>>>(x, wih, b_ih, g_ih, be_ih, ziLn);

    // 32K h + 2x32K z (double buffer) + 256B stats = 98,560 B of LDS
    const size_t smem = (size_t)(3 * BDIM * HDIM * 2 + 64 * 4);
    (void)hipFuncSetAttribute((const void*)rnn_rec_pre,
                              hipFuncAttributeMaxDynamicSharedMemorySize, (int)smem);
    rnn_rec_pre<<<LNUM, 1024, smem, stream>>>(ziLn, whh, b_hh, g_hh, be_hh, h3, out);
  } else {
    // Fallback: fused recurrence.
    const size_t smem = (size_t)(2 * BDIM * HDIM * 2 + 128 * 4); // 66,048 B
    (void)hipFuncSetAttribute((const void*)rnn_rec_fused,
                              hipFuncAttributeMaxDynamicSharedMemorySize, (int)smem);
    rnn_rec_fused<<<LNUM, 1024, smem, stream>>>(x, wih, b_ih, g_ih, be_ih,
                                                whh, b_hh, g_hh, be_hh, h3, out);
  }

  out_gemm<<<(TDIM * BDIM) / 16, 256, 0, stream>>>(h3, who, b_ho, x, out);
}
